// MS_Block_Conv_MoE_84172769067793
// MI455X (gfx1250) — matrix operand order repaired and checked by reading, not compiled, not run on hardware
//
#include <hip/hip_runtime.h>

// ---------------------------------------------------------------------------
// MS_Block_Conv_MoE on MI455X (gfx1250, wave32, WMMA).
//
// Roofline: effective GEMM work after top-2 gating = B*K (=32) sample-expert
// pairs * ~1.07 GFLOP = ~34 GFLOP, all in v_wmma_f32_16x16x32_f16 (spikes are
// exactly {0,1} -> exact in f16; f32 accumulate). Working set (~80 MB) fits in
// the 192 MB L2. Per-wave 32x64 D tiles: each B fragment feeds 2 WMMAs
// (32 flops/L2-byte on the weight stream) to keep the WMMA pipes fed.
// ---------------------------------------------------------------------------

typedef __attribute__((ext_vector_type(16))) _Float16 v16h;
typedef __attribute__((ext_vector_type(8)))  float    v8f;

#define T_   4
#define B_   16
#define C_   256
#define HW_  256
#define HID_ 1024
#define E_   8
#define M_   (T_ * B_ * HW_)   // 16384 (m = pixel*4 + t)
#define EPS_ 1e-5f

// ---------------- out = x (residual init) ----------------------------------
__global__ void k_copy_out(const float4* __restrict__ x, float4* __restrict__ out, int n4) {
  int i = blockIdx.x * blockDim.x + threadIdx.x;
  if (i < n4) out[i] = x[i];
}

// ---------------- weights f32 -> f16 (both fc1 and fc2, same count) --------
__global__ void k_cvt_weights(const float* __restrict__ w1, const float* __restrict__ w2,
                              _Float16* __restrict__ w1h, _Float16* __restrict__ w2h, int n) {
  int i = blockIdx.x * blockDim.x + threadIdx.x;
  if (i < n) {
    w1h[i] = (_Float16)w1[i];
    w2h[i] = (_Float16)w2[i];
  }
}

// ---------------- xbar[b,c] = mean over (t,hw) of x -------------------------
__global__ void k_xbar(const float* __restrict__ x, float* __restrict__ xbar) {
  int bc = blockIdx.x;                 // b*C + c
  int b = bc >> 8, c = bc & (C_ - 1);
  int tid = threadIdx.x;               // hw
  float s = 0.f;
#pragma unroll
  for (int t = 0; t < T_; ++t)
    s += x[(((size_t)t * B_ + b) * C_ + c) * HW_ + tid];
  __shared__ float red[256];
  red[tid] = s;
  __syncthreads();
  for (int off = 128; off > 0; off >>= 1) {
    if (tid < off) red[tid] += red[tid + off];
    __syncthreads();
  }
  if (tid == 0) xbar[bc] = red[0] * (1.0f / (T_ * HW_));
}

// ---------------- router: logits -> softmax -> top2 -> dense gate weights --
__global__ void k_router(const float* __restrict__ xbar, const float* __restrict__ rw,
                         const float* __restrict__ rb, const float* __restrict__ g,
                         const float* __restrict__ be, const float* __restrict__ mean,
                         const float* __restrict__ var, float* __restrict__ dw) {
  __shared__ float lg[B_][E_];
  int i = threadIdx.x;
  if (i < B_ * E_) {
    int b = i / E_, e = i % E_;
    float acc = 0.f;
    for (int c = 0; c < C_; ++c) acc += xbar[b * C_ + c] * rw[e * C_ + c];
    acc += rb[e];
    float sc = g[e] * rsqrtf(var[e] + EPS_);
    lg[b][e] = (acc - mean[e]) * sc + be[e];
  }
  __syncthreads();
  if (i < B_) {
    float mx = lg[i][0];
    for (int e = 1; e < E_; ++e) mx = fmaxf(mx, lg[i][e]);
    float p[E_];
    for (int e = 0; e < E_; ++e) p[e] = __expf(lg[i][e] - mx);
    int i1 = 0;
    for (int e = 1; e < E_; ++e) if (p[e] > p[i1]) i1 = e;        // top-1, lowest idx wins ties
    int i2 = (i1 == 0) ? 1 : 0;
    for (int e = 0; e < E_; ++e) if (e != i1 && p[e] > p[i2]) i2 = e;
    float s12 = p[i1] + p[i2];                                     // softmax denom cancels
    for (int e = 0; e < E_; ++e) dw[i * E_ + e] = 0.f;
    dw[i * E_ + i1] = p[i1] / s12;
    dw[i * E_ + i2] = p[i2] / s12;
  }
}

// ---------------- LIF over x -> binary spikes s1 (f16 A-matrix, [m][C]) ----
__global__ void k_spike1(const float* __restrict__ x, _Float16* __restrict__ s1,
                         float itau, const float* __restrict__ dw, int e) {
  int idx = blockIdx.x * blockDim.x + threadIdx.x;  // b*C*HW + c*HW + hw
  int hw = idx & (HW_ - 1);
  int c  = (idx >> 8) & (C_ - 1);
  int b  = idx >> 16;
  if (dw[b * E_ + e] == 0.f) return;                // expert not routed for this sample
  int p = b * HW_ + hw;
  float v = 0.f;
#pragma unroll
  for (int t = 0; t < T_; ++t) {
    float xv = x[(((size_t)t * B_ + b) * C_ + c) * HW_ + hw];
    v += (xv - v) * itau;
    float s = (v >= 1.0f) ? 1.0f : 0.0f;
    v *= (1.0f - s);
    s1[(size_t)(p * T_ + t) * C_ + c] = (_Float16)s;
  }
}

// ------ fragment loaders (ISA §7.12.2 layouts) ------------------------------
__device__ __forceinline__ v16h load_afrag(const _Float16* arow, int k0, int grp) {
  // 16-bit A 16x32: lane holds 8-half chunks at k0+grp*8 and k0+16+grp*8
  union { v16h v; uint4 u[2]; } f;
  f.u[0] = *reinterpret_cast<const uint4*>(arow + k0 + grp * 8);
  f.u[1] = *reinterpret_cast<const uint4*>(arow + k0 + 16 + grp * 8);
  return f.v;
}
__device__ __forceinline__ v16h load_bfrag(const _Float16* bcol, int k0, int grp) {
  // 16-bit B 32x16: lane n holds 8-half chunks at k0+grp*8 and k0+16+grp*8
  union { v16h v; uint4 u[2]; } f;
  const _Float16* p = bcol + k0 + grp * 8;
  f.u[0] = *reinterpret_cast<const uint4*>(p);
  f.u[1] = *reinterpret_cast<const uint4*>(p + 16);
  return f.v;
}

// ---------------- GEMM1 (M=16384,N=1024,K=256) + BN1 + LIF2 -> s2 ----------
// Block: 8 waves, 32-row A strip in LDS. Each wave owns a 32x64 D tile
// (2x4 wmma accumulators); each B fragment feeds 2 WMMAs.
__global__ void __launch_bounds__(256)
k_gemm1_lif(const _Float16* __restrict__ s1, const _Float16* __restrict__ w1h,
            const float* __restrict__ b1, const float* __restrict__ g1,
            const float* __restrict__ be1, const float* __restrict__ m1,
            const float* __restrict__ v1, _Float16* __restrict__ s2,
            float itau, const float* __restrict__ dw, int e) {
  int m32 = blockIdx.x * 32;
  int b = m32 >> 10;                          // sample index: uniform over the block
  if (dw[b * E_ + e] == 0.f) return;          // skip unrouted sample entirely

  __shared__ __align__(16) _Float16 ldsA[32 * C_];   // 16 KB
  {
    const uint4* src = reinterpret_cast<const uint4*>(s1 + (size_t)m32 * C_);
    uint4* dst = reinterpret_cast<uint4*>(ldsA);
    for (int j = threadIdx.x; j < (32 * C_) / 8; j += 256) dst[j] = src[j];
  }
  __syncthreads();

  int wave = threadIdx.x >> 5, lane = threadIdx.x & 31;
  int row = lane & 15, grp = lane >> 4;
  int n0 = (blockIdx.y * 8 + wave) * 64;
  v8f acc[2][4] = {};
  const _Float16* arow0 = ldsA + row * C_;          // M-subtile 0: rows 0..15
  const _Float16* arow1 = ldsA + (row + 16) * C_;   // M-subtile 1: rows 16..31

  for (int k0 = 0; k0 < C_; k0 += 32) {
    v16h af0 = load_afrag(arow0, k0, grp);
    v16h af1 = load_afrag(arow1, k0, grp);
#pragma unroll
    for (int nt = 0; nt < 4; ++nt) {
      v16h bf = load_bfrag(w1h + (size_t)(n0 + nt * 16 + row) * C_, k0, grp);
      acc[0][nt] = __builtin_amdgcn_wmma_f32_16x16x32_f16(false, af0, false, bf,
                                                          (short)0, acc[0][nt], false, false);
      acc[1][nt] = __builtin_amdgcn_wmma_f32_16x16x32_f16(false, af1, false, bf,
                                                          (short)0, acc[1][nt], false, false);
    }
  }

  // Epilogue: each lane's 8 acc rows = full T=4 series of 2 pixels -> fused LIF scan.
#pragma unroll
  for (int nt = 0; nt < 4; ++nt) {
    int o = n0 + nt * 16 + row;
    float sc = g1[o] * rsqrtf(v1[o] + EPS_);
    float sh = be1[o] + (b1[o] - m1[o]) * sc;
#pragma unroll
    for (int mt = 0; mt < 2; ++mt) {
      int mbase = m32 + mt * 16 + grp * 8;
#pragma unroll
      for (int pix = 0; pix < 2; ++pix) {
        float v = 0.f;
#pragma unroll
        for (int t = 0; t < T_; ++t) {
          float h = acc[mt][nt][pix * 4 + t] * sc + sh;
          v += (h - v) * itau;
          float s = (v >= 1.0f) ? 1.0f : 0.0f;
          v *= (1.0f - s);
          s2[(size_t)(mbase + pix * 4 + t) * HID_ + o] = (_Float16)s;
        }
      }
    }
  }
}

// ---------------- GEMM2 (M=16384,N=256,K=1024) + BN2 + gated accumulate ----
__global__ void __launch_bounds__(128)
k_gemm2_out(const _Float16* __restrict__ s2, const _Float16* __restrict__ w2h,
            const float* __restrict__ b2, const float* __restrict__ g2,
            const float* __restrict__ be2, const float* __restrict__ m2,
            const float* __restrict__ v2, const float* __restrict__ dw,
            int e, float* __restrict__ out) {
  int m32 = blockIdx.x * 32;
  int b = m32 >> 10;
  float dwv = dw[b * E_ + e];
  if (dwv == 0.f) return;

  __shared__ __align__(16) _Float16 ldsA[32 * HID_];   // 64 KB (<< 320 KB/WGP)
  {
    const uint4* src = reinterpret_cast<const uint4*>(s2 + (size_t)m32 * HID_);
    uint4* dst = reinterpret_cast<uint4*>(ldsA);
    for (int j = threadIdx.x; j < (32 * HID_) / 8; j += 128) dst[j] = src[j];
  }
  __syncthreads();

  int wave = threadIdx.x >> 5, lane = threadIdx.x & 31;
  int row = lane & 15, grp = lane >> 4;
  int n0 = wave * 64;                          // 4 waves cover N=256
  v8f acc[2][4] = {};
  const _Float16* arow0 = ldsA + row * HID_;
  const _Float16* arow1 = ldsA + (row + 16) * HID_;

  for (int k0 = 0; k0 < HID_; k0 += 32) {
    v16h af0 = load_afrag(arow0, k0, grp);
    v16h af1 = load_afrag(arow1, k0, grp);
#pragma unroll
    for (int nt = 0; nt < 4; ++nt) {
      v16h bf = load_bfrag(w2h + (size_t)(n0 + nt * 16 + row) * HID_, k0, grp);
      acc[0][nt] = __builtin_amdgcn_wmma_f32_16x16x32_f16(false, af0, false, bf,
                                                          (short)0, acc[0][nt], false, false);
      acc[1][nt] = __builtin_amdgcn_wmma_f32_16x16x32_f16(false, af1, false, bf,
                                                          (short)0, acc[1][nt], false, false);
    }
  }

#pragma unroll
  for (int nt = 0; nt < 4; ++nt) {
    int c = n0 + nt * 16 + row;
    float sc = g2[c] * rsqrtf(v2[c] + EPS_);
    float sh = be2[c] + (b2[c] - m2[c]) * sc;
#pragma unroll
    for (int mt = 0; mt < 2; ++mt) {
      int mbase = m32 + mt * 16 + grp * 8;
#pragma unroll
      for (int r = 0; r < 8; ++r) {
        int m = mbase + r;
        int t = m & 3, p = m >> 2;
        int bb = p >> 8, hw = p & 255;
        size_t oi = (((size_t)t * B_ + bb) * C_ + c) * HW_ + hw;
        out[oi] += dwv * (acc[mt][nt][r] * sc + sh);   // single writer per element per launch
      }
    }
  }
}

// ---------------------------------------------------------------------------
extern "C" void kernel_launch(void* const* d_in, const int* in_sizes, int n_in,
                              void* d_out, int out_size, void* d_ws, size_t ws_size,
                              hipStream_t stream) {
  const float* x   = (const float*)d_in[0];
  const float* rw  = (const float*)d_in[1];
  const float* rb  = (const float*)d_in[2];
  const float* rg  = (const float*)d_in[3];
  const float* rbe = (const float*)d_in[4];
  const float* rm  = (const float*)d_in[5];
  const float* rv  = (const float*)d_in[6];
  const float* w1  = (const float*)d_in[7];
  const float* b1  = (const float*)d_in[8];
  const float* g1  = (const float*)d_in[9];
  const float* be1 = (const float*)d_in[10];
  const float* m1  = (const float*)d_in[11];
  const float* v1  = (const float*)d_in[12];
  const float* w2  = (const float*)d_in[13];
  const float* b2  = (const float*)d_in[14];
  const float* g2  = (const float*)d_in[15];
  const float* be2 = (const float*)d_in[16];
  const float* m2  = (const float*)d_in[17];
  const float* v2  = (const float*)d_in[18];
  float* out = (float*)d_out;

  // workspace carve-up (~48 MB total)
  char* ws = (char*)d_ws;
  float*    dwv  = (float*)ws;    ws += 4096;                          // (B,E) gate weights
  float*    xbar = (float*)ws;    ws += (size_t)B_ * C_ * 4;           // 16 KB
  _Float16* w1h  = (_Float16*)ws; ws += (size_t)E_ * HID_ * C_ * 2;    // 4 MB
  _Float16* w2h  = (_Float16*)ws; ws += (size_t)E_ * C_ * HID_ * 2;    // 4 MB
  _Float16* s1   = (_Float16*)ws; ws += (size_t)M_ * C_ * 2;           // 8 MB
  _Float16* s2   = (_Float16*)ws;                                      // 32 MB

  // 1) residual init: out = x
  int n4 = (T_ * B_ * C_ * HW_) / 4;
  k_copy_out<<<n4 / 256, 256, 0, stream>>>((const float4*)x, (float4*)out, n4);

  // 2) f16 weights for WMMA
  int nw = E_ * HID_ * C_;
  k_cvt_weights<<<nw / 256, 256, 0, stream>>>(w1, w2, w1h, w2h, nw);

  // 3) router
  k_xbar<<<B_ * C_, 256, 0, stream>>>(x, xbar);
  k_router<<<1, 128, 0, stream>>>(xbar, rw, rb, rg, rbe, rm, rv, dwv);

  // 4) experts (sequential launches; gated blocks early-exit => ~4x less work)
  for (int e = 0; e < E_; ++e) {
    float tau  = 1.5f + (float)e * (2.5f / 7.0f);   // linspace(1.5, 4.0, 8)
    float itau = 1.0f / tau;
    k_spike1<<<(B_ * C_ * HW_) / 256, 256, 0, stream>>>(x, s1, itau, dwv, e);
    k_gemm1_lif<<<dim3(M_ / 32, 2), 256, 0, stream>>>(
        s1, w1h + (size_t)e * HID_ * C_,
        b1 + e * HID_, g1 + e * HID_, be1 + e * HID_, m1 + e * HID_, v1 + e * HID_,
        s2, itau, dwv, e);
    k_gemm2_out<<<M_ / 32, 128, 0, stream>>>(
        s2, w2h + (size_t)e * C_ * HID_,
        b2 + e * C_, g2 + e * C_, be2 + e * C_, m2 + e * C_, v2 + e * C_,
        dwv, e, out);
  }
}
